// DecoderGNN_88888643158446
// MI455X (gfx1250) — matrix operand order repaired and checked by reading, not compiled, not run on hardware
//
#include <hip/hip_runtime.h>
#include <hip/hip_bf16.h>

// ---------------- problem constants (match reference) ----------------
#define NBATCH   4
#define NNODE    50000
#define BN       (NBATCH * NNODE)      // 200000 flat nodes
#define NEDGE    150000
#define BE       (NBATCH * NEDGE)      // 600000 flat edges
#define CODE     64
#define HID      128
#define OUTD     64
#define NLAYER   3
#define LNEPS    1e-5f

// ---------------- workspace layout ----------------
// h   : BN*HID f32  (102,400,000 B)
// agg : BN*HID f32  (102,400,000 B)
// wbf : bf16 weights, stored TRANSPOSED ([col][k]) for vector B-frag loads
#define H_BYTES   ((size_t)BN * HID * 4)
#define WIN_OFF   0        // W_in^T  : [128][64]
#define W1_OFF    8192     // W1^T    : 3 x [128][128]
#define W2_OFF    57344    // W2^T    : 3 x [128][128]
#define WO1_OFF   106496   // W_o1^T  : [128][128]
#define WO2_OFF   122880   // W_o2^T  : [64][128]
#define WTOT      131072

typedef __attribute__((ext_vector_type(16))) __bf16 bf16x16;
typedef __attribute__((ext_vector_type(8)))  __bf16 bf16x8;
typedef __attribute__((ext_vector_type(8)))  float  f32x8;
typedef __attribute__((ext_vector_type(4)))  unsigned int u32x4;
typedef __attribute__((ext_vector_type(8)))  int    i32x8;
typedef __attribute__((ext_vector_type(4)))  int    i32x4;

// TDM builtin arity differs between toolchains (probe-verified):
//   ROCm 7.2 / clang-22 : 5 args (g0, g1, g2, g3, cpol)
//   amdgpu-toolchain / clang-23+ : 6 args (g0, g1, g2, g3, i32x8, cpol)
#if __has_builtin(__builtin_amdgcn_tensor_load_to_lds)
  #if defined(__clang_major__) && (__clang_major__ >= 23)
    #define TDM_ARITY 6
  #else
    #define TDM_ARITY 5
  #endif
#else
  #define TDM_ARITY 0
#endif

// ---------------- WMMA helpers ----------------
__device__ __forceinline__ f32x8 wmma_bf16(bf16x16 a, bf16x16 b, f32x8 c) {
  return __builtin_amdgcn_wmma_f32_16x16x32_bf16(false, a, false, b,
                                                 (short)0, c, false, false);
}

// A fragment: 16x32 bf16 from row-major LDS tile (element stride `stride`).
// ISA 7.12.2 layout: lane m holds row m; lanes 0-15 carry K = kb+0..7 and
// kb+16..23, lanes 16-31 carry K = kb+8..15 and kb+24..31 -> per lane this is
// two contiguous 16B runs => two ds_load_b128.
__device__ __forceinline__ bf16x16 load_a_frag(const __bf16* s, int stride,
                                               int row0, int kbase, int lane) {
  int hh = lane >> 4;
  const __bf16* p = s + (row0 + (lane & 15)) * stride + kbase + (hh << 3);
  bf16x8 lo = *(const bf16x8*)(p);       // K = kbase + hh*8 + 0..7
  bf16x8 hi = *(const bf16x8*)(p + 16);  // K = kbase + 16 + hh*8 + 0..7
  return __builtin_shufflevector(lo, hi, 0, 1, 2, 3, 4, 5, 6, 7,
                                 8, 9, 10, 11, 12, 13, 14, 15);
}

// B fragment from TRANSPOSED LDS weight tile sWt[col][k] (stride elements).
// Lanes 0-15 carry K = kb+0..7 and kb+16..23 of column n, lanes 16-31 carry
// K = kb+8..15 and kb+24..31 -> two contiguous 16B runs => two ds_load_b128.
__device__ __forceinline__ bf16x16 load_b_frag(const __bf16* s, int stride,
                                               int col0, int kbase, int lane) {
  int hh = lane >> 4;
  const __bf16* p = s + (col0 + (lane & 15)) * stride + kbase + (hh << 3);
  bf16x8 lo = *(const bf16x8*)(p);
  bf16x8 hi = *(const bf16x8*)(p + 16);
  return __builtin_shufflevector(lo, hi, 0, 1, 2, 3, 4, 5, 6, 7,
                                 8, 9, 10, 11, 12, 13, 14, 15);
}

// ---------------- weight staging: TDM (CDNA5 tensor DMA) or manual ----------------
// Flat copy of a rows x cols bf16 row-major matrix (already transposed in ws)
// into LDS with row stride ldsStride (elements). TDM D# pad_enable inserts
// the per-row padding during the DMA.
__device__ __forceinline__ void stage_weight_tile(__bf16* dstLds,
                                                  const __bf16* srcGlb,
                                                  int rows, int cols,
                                                  int ldsStride, int tid) {
#if TDM_ARITY
  if ((tid >> 5) == 0) {  // one wave issues the DMA (EXEC ignored by TDM)
    unsigned long long ga = (unsigned long long)(const void*)srcGlb;
    unsigned int la = (unsigned int)(unsigned long long)(void*)dstLds;
    int rowDwords = cols >> 1;                       // bf16 row in DWORDs
    int intervalCode = 31 - __builtin_clz((unsigned)rowDwords) - 1; // 2^(c+1) dw
    int padDwords = (ldsStride - cols) >> 1;         // pad per row in DWORDs
    u32x4 g0; i32x8 g1; i32x4 g2 = {0, 0, 0, 0}; i32x4 g3 = {0, 0, 0, 0};
    g0[0] = 1u;                                      // count=1, no gather
    g0[1] = la;                                      // lds_addr
    g0[2] = (unsigned int)(ga & 0xFFFFFFFFull);      // global_addr lo
    g0[3] = (unsigned int)((ga >> 32) & 0x01FFFFFFull) | 0x80000000u; // type=2
    g1[0] = (1 << 16)                                // data_size = 2B
          | (1 << 20)                                // pad_enable
          | (intervalCode << 22)                     // pad_interval
          | ((padDwords - 1) << 25);                 // pad_amount
    g1[1] = (cols & 0xFFFF) << 16;                   // tensor_dim0 lo16
    g1[2] = (cols >> 16) | ((rows & 0xFFFF) << 16);  // dim0 hi | dim1 lo
    g1[3] = (rows >> 16) | (cols << 16);             // dim1 hi | tile_dim0
    g1[4] = rows;                                    // tile_dim1 (tile_dim2=0)
    g1[5] = cols;                                    // tensor_dim0_stride lo
    g1[6] = 0; g1[7] = 0;
#if TDM_ARITY == 5
    __builtin_amdgcn_tensor_load_to_lds(g0, g1, g2, g3, 0);
#else
    i32x8 g4 = {0, 0, 0, 0, 0, 0, 0, 0};
    __builtin_amdgcn_tensor_load_to_lds(g0, g1, g2, g3, g4, 0);
#endif
    __builtin_amdgcn_s_wait_tensorcnt(0);
  }
#else
  int total = rows * cols;
  for (int i = tid; i < total; i += 256) {
    int r = i / cols, c = i - r * cols;
    dstLds[r * ldsStride + c] = srcGlb[i];
  }
#endif
}

// ---------------- kernel 1: fp32 -> bf16 weight conversion + transpose ----------------
__global__ void cvt_weights(const float* __restrict__ Win,
                            const float* __restrict__ W1,
                            const float* __restrict__ W2,
                            const float* __restrict__ Wo1,
                            const float* __restrict__ Wo2,
                            __bf16* __restrict__ dst) {
  int i = blockIdx.x * blockDim.x + threadIdx.x;
  if (i >= WTOT) return;
  const float* src;
  int off, j, rows, cols;
  if (i < W1_OFF) {                // W_in: 64 x 128
    off = WIN_OFF; j = i - off; src = Win; rows = CODE; cols = HID;
  } else if (i < W2_OFF) {         // W1[l]: 128 x 128
    int li = (i - W1_OFF) / (HID * HID);
    off = W1_OFF + li * HID * HID; j = i - off;
    src = W1 + li * HID * HID; rows = HID; cols = HID;
  } else if (i < WO1_OFF) {        // W2[l]: 128 x 128
    int li = (i - W2_OFF) / (HID * HID);
    off = W2_OFF + li * HID * HID; j = i - off;
    src = W2 + li * HID * HID; rows = HID; cols = HID;
  } else if (i < WO2_OFF) {        // W_o1: 128 x 128
    off = WO1_OFF; j = i - off; src = Wo1; rows = HID; cols = HID;
  } else {                         // W_o2: 128 x 64
    off = WO2_OFF; j = i - off; src = Wo2; rows = HID; cols = OUTD;
  }
  int k = j / cols, c = j - k * cols;
  dst[off + c * rows + k] = (__bf16)src[j];  // transposed store: [col][k]
}

// ---------------- kernel 2: input projection h = z_q @ W_in + b_in ----------------
__global__ __launch_bounds__(256) void input_proj(const float* __restrict__ zq,
                                                  const __bf16* __restrict__ wbf,
                                                  const float* __restrict__ bin,
                                                  float* __restrict__ h) {
  __shared__ __align__(16) __bf16 sA[64 * 72];    // z_q tile: 64 nodes x 64 (pad 8)
  __shared__ __align__(16) __bf16 sB[128 * 72];   // W_in^T: [128 col][64 k] (pad 8)
  __shared__ float sb[HID];
  int t = threadIdx.x;
  int node0 = blockIdx.x * 64;
  for (int i = t; i < 64 * 64; i += 256) {
    int r = i >> 6, c = i & 63;
    sA[r * 72 + c] = (__bf16)zq[(size_t)(node0 + r) * CODE + c];
  }
  stage_weight_tile(sB, wbf, HID, CODE, 72, t);   // transposed: 128 rows of 64
  if (t < HID) sb[t] = bin[t];
  __syncthreads();

  int w = t >> 5, lane = t & 31, hh = lane >> 4, l15 = lane & 15;
  int row0 = (w >> 1) * 16;
  bf16x16 aF[2];
#pragma unroll
  for (int kb = 0; kb < 2; ++kb) aF[kb] = load_a_frag(sA, 72, row0, kb * 32, lane);

  for (int nt = (w & 1) * 4; nt < (w & 1) * 4 + 4; ++nt) {
    int col0 = nt * 16;
    f32x8 acc = {};
#pragma unroll
    for (int kb = 0; kb < 2; ++kb)
      acc = wmma_bf16(aF[kb], load_b_frag(sB, 72, col0, kb * 32, lane), acc);
    int col = col0 + l15;
#pragma unroll
    for (int r = 0; r < 8; ++r) {
      int grow = node0 + row0 + r + hh * 8;
      h[(size_t)grow * HID + col] = acc[r] + sb[col];
    }
  }
}

// ---------------- kernel 3: GIN aggregation agg[dst] += h[src] ----------------
#define EPW 4  // edges per wave (NEDGE % EPW == 0, so no batch straddle)
__global__ __launch_bounds__(256) void gin_gather(const long long* __restrict__ ei,
                                                  const float* __restrict__ h,
                                                  float* __restrict__ agg) {
  int wid = blockIdx.x * (blockDim.x >> 5) + (threadIdx.x >> 5);
  int lane = threadIdx.x & 31;
  int e0 = wid * EPW;
  if (e0 >= BE) return;
  int b = e0 / NEDGE;
  int eb = e0 - b * NEDGE;
  const long long* p = ei + (size_t)b * 2 * NEDGE;
  int s[EPW], d[EPW];
#pragma unroll
  for (int i = 0; i < EPW; ++i) {
    s[i] = (int)p[eb + i] + b * NNODE;
    d[i] = (int)p[NEDGE + eb + i] + b * NNODE;
  }
#pragma unroll
  for (int i = 0; i < EPW; ++i)  // global_prefetch_b8: hide random-row latency
    __builtin_prefetch(h + (size_t)s[i] * HID + lane * 4, 0, 0);
#pragma unroll
  for (int i = 0; i < EPW; ++i) {
    float4 v = *(const float4*)(h + (size_t)s[i] * HID + lane * 4);
    float* a = agg + (size_t)d[i] * HID + lane * 4;
    atomicAdd(a + 0, v.x);
    atomicAdd(a + 1, v.y);
    atomicAdd(a + 2, v.z);
    atomicAdd(a + 3, v.w);
  }
}

// ---------------- kernel 4: fused GIN layer (MLP + LN + ReLU + residual) ----------------
__global__ __launch_bounds__(256) void gin_layer(float* h,  // in-place update
                                                 const float* __restrict__ agg,
                                                 const __bf16* __restrict__ w1bf,
                                                 const __bf16* __restrict__ w2bf,
                                                 const float* __restrict__ b1,
                                                 const float* __restrict__ b2,
                                                 const float* __restrict__ gam,
                                                 const float* __restrict__ bet) {
  __shared__ __align__(16) char smem[8704 + 34816 + 8704];  // 52224 B
  __bf16* sX = (__bf16*)smem;                  // x tile  : [32][136]
  __bf16* sW = (__bf16*)(smem + 8704);         // W^T     : [128][136]
  __bf16* sY = (__bf16*)(smem + 8704 + 34816); // y tile  : [32][136]
  float*  sZ = (float*)smem;                   // [32][132] overlays dead sX/sW
  __shared__ float sb1[HID], sb2[HID], sg[HID], sbt[HID];
  __shared__ float sMu[32], sRs[32];

  int t = threadIdx.x;
  int node0 = blockIdx.x * 32;
  float res[16];  // residual h values cached in registers (16 = 32*128/256)
#pragma unroll
  for (int ii = 0; ii < 16; ++ii) {
    int i = t + ii * 256;
    int r = i >> 7, c = i & 127;
    size_t gi = (size_t)(node0 + r) * HID + c;
    float hv = h[gi];
    res[ii] = hv;
    sX[r * 136 + c] = (__bf16)(hv + agg[gi]);  // x = h + segment_sum
  }
  if (t < HID) { sb1[t] = b1[t]; sb2[t] = b2[t]; sg[t] = gam[t]; sbt[t] = bet[t]; }
  stage_weight_tile(sW, w1bf, HID, HID, 136, t);
  __syncthreads();

  int w = t >> 5, lane = t & 31, hh = lane >> 4, l15 = lane & 15;
  int row0 = (w >> 2) * 16;
  int ntb = (w & 3) * 2;

  // GEMM1 + bias + ReLU -> sY (bf16); A fragments loaded once, reused per tile
  {
    bf16x16 aF[4];
#pragma unroll
    for (int kb = 0; kb < 4; ++kb) aF[kb] = load_a_frag(sX, 136, row0, kb * 32, lane);
    for (int ti = 0; ti < 2; ++ti) {
      int col0 = (ntb + ti) * 16;
      f32x8 acc = {};
#pragma unroll
      for (int kb = 0; kb < 4; ++kb)
        acc = wmma_bf16(aF[kb], load_b_frag(sW, 136, col0, kb * 32, lane), acc);
      int col = col0 + l15;
#pragma unroll
      for (int r = 0; r < 8; ++r) {
        float v = acc[r] + sb1[col];
        sY[(row0 + r + hh * 8) * 136 + col] = (__bf16)fmaxf(v, 0.f);
      }
    }
  }
  __syncthreads();
  stage_weight_tile(sW, w2bf, HID, HID, 136, t);
  __syncthreads();

  // GEMM2 into registers
  f32x8 acc2[2];
  {
    bf16x16 aF[4];
#pragma unroll
    for (int kb = 0; kb < 4; ++kb) aF[kb] = load_a_frag(sY, 136, row0, kb * 32, lane);
    for (int ti = 0; ti < 2; ++ti) {
      int col0 = (ntb + ti) * 16;
      f32x8 acc = {};
#pragma unroll
      for (int kb = 0; kb < 4; ++kb)
        acc = wmma_bf16(aF[kb], load_b_frag(sW, 136, col0, kb * 32, lane), acc);
      acc2[ti] = acc;
    }
  }
  __syncthreads();  // sX/sW/sY now dead -> safe to overlay sZ

  for (int ti = 0; ti < 2; ++ti) {
    int col = (ntb + ti) * 16 + l15;
#pragma unroll
    for (int r = 0; r < 8; ++r)
      sZ[(row0 + r + hh * 8) * 132 + col] = acc2[ti][r] + sb2[col];
  }
  __syncthreads();

  if (t < 32) {  // per-node LayerNorm stats
    float s = 0.f, s2 = 0.f;
    for (int c = 0; c < HID; ++c) { float v = sZ[t * 132 + c]; s += v; s2 += v * v; }
    float mu = s * (1.f / HID);
    float var = s2 * (1.f / HID) - mu * mu;
    sMu[t] = mu;
    sRs[t] = rsqrtf(var + LNEPS);
  }
  __syncthreads();

#pragma unroll
  for (int ii = 0; ii < 16; ++ii) {
    int i = t + ii * 256;
    int r = i >> 7, c = i & 127;
    size_t gi = (size_t)(node0 + r) * HID + c;
    float x = (sZ[r * 132 + c] - sMu[r]) * sRs[r] * sg[c] + sbt[c];
    h[gi] = res[ii] + fmaxf(x, 0.f);  // residual + ReLU (res cached in regs)
  }
}

// ---------------- kernel 5: output projection ----------------
__global__ __launch_bounds__(256) void out_proj(const float* __restrict__ h,
                                                const __bf16* __restrict__ wo1bf,
                                                const __bf16* __restrict__ wo2bf,
                                                const float* __restrict__ bo1,
                                                const float* __restrict__ bo2,
                                                float* __restrict__ out) {
  __shared__ __align__(16) __bf16 sX[32 * 136];
  __shared__ __align__(16) __bf16 sW[128 * 136];
  __shared__ __align__(16) __bf16 sY[32 * 136];
  __shared__ float sb1[HID], sb2[OUTD];
  int t = threadIdx.x;
  int node0 = blockIdx.x * 32;
  for (int i = t; i < 32 * HID; i += 256) {
    int r = i >> 7, c = i & 127;
    sX[r * 136 + c] = (__bf16)h[(size_t)(node0 + r) * HID + c];
  }
  if (t < HID)  sb1[t] = bo1[t];
  if (t < OUTD) sb2[t] = bo2[t];
  stage_weight_tile(sW, wo1bf, HID, HID, 136, t);   // W_o1^T [128][128]
  __syncthreads();

  int w = t >> 5, lane = t & 31, hh = lane >> 4, l15 = lane & 15;
  int row0 = (w >> 2) * 16;
  int ntb = (w & 3) * 2;

  // GEMM1 + bias + ReLU -> sY
  {
    bf16x16 aF[4];
#pragma unroll
    for (int kb = 0; kb < 4; ++kb) aF[kb] = load_a_frag(sX, 136, row0, kb * 32, lane);
    for (int ti = 0; ti < 2; ++ti) {
      int col0 = (ntb + ti) * 16;
      f32x8 acc = {};
#pragma unroll
      for (int kb = 0; kb < 4; ++kb)
        acc = wmma_bf16(aF[kb], load_b_frag(sW, 136, col0, kb * 32, lane), acc);
      int col = col0 + l15;
#pragma unroll
      for (int r = 0; r < 8; ++r) {
        float v = acc[r] + sb1[col];
        sY[(row0 + r + hh * 8) * 136 + col] = (__bf16)fmaxf(v, 0.f);
      }
    }
  }
  __syncthreads();
  stage_weight_tile(sW, wo2bf, OUTD, HID, 136, t);  // W_o2^T [64][128]
  __syncthreads();

  // GEMM2: 2 M-tiles x 4 N-tiles = 8 tiles, one per wave
  int col0 = (w & 3) * 16;
  f32x8 acc = {};
#pragma unroll
  for (int kb = 0; kb < 4; ++kb)
    acc = wmma_bf16(load_a_frag(sY, 136, row0, kb * 32, lane),
                    load_b_frag(sW, 136, col0, kb * 32, lane), acc);
  int col = col0 + l15;
#pragma unroll
  for (int r = 0; r < 8; ++r) {
    int grow = node0 + row0 + r + hh * 8;
    out[(size_t)grow * OUTD + col] = acc[r] + sb2[col];
  }
}

// ---------------- host launcher ----------------
extern "C" void kernel_launch(void* const* d_in, const int* in_sizes, int n_in,
                              void* d_out, int out_size, void* d_ws, size_t ws_size,
                              hipStream_t stream) {
  const float*     zq   = (const float*)d_in[0];
  const long long* ei   = (const long long*)d_in[1];
  const float*     Win  = (const float*)d_in[2];
  const float*     bin  = (const float*)d_in[3];
  const float*     W1   = (const float*)d_in[4];
  const float*     b1   = (const float*)d_in[5];
  const float*     W2   = (const float*)d_in[6];
  const float*     b2   = (const float*)d_in[7];
  const float*     gam  = (const float*)d_in[8];
  const float*     bet  = (const float*)d_in[9];
  const float*     Wo1  = (const float*)d_in[10];
  const float*     bo1  = (const float*)d_in[11];
  const float*     Wo2  = (const float*)d_in[12];
  const float*     bo2  = (const float*)d_in[13];
  float* out = (float*)d_out;

  char* ws = (char*)d_ws;
  float*  h   = (float*)ws;                       // BN*HID f32
  float*  agg = (float*)(ws + H_BYTES);           // BN*HID f32
  __bf16* wbf = (__bf16*)(ws + 2 * H_BYTES);      // 131072 bf16 (transposed)

  cvt_weights<<<(WTOT + 255) / 256, 256, 0, stream>>>(Win, W1, W2, Wo1, Wo2, wbf);
  input_proj<<<BN / 64, 256, 0, stream>>>(zq, wbf + WIN_OFF, bin, h);

  for (int l = 0; l < NLAYER; ++l) {
    hipMemsetAsync(agg, 0, H_BYTES, stream);
    gin_gather<<<(BE / EPW) / 8, 256, 0, stream>>>(ei, h, agg);
    gin_layer<<<BN / 32, 256, 0, stream>>>(h, agg,
                                           wbf + W1_OFF + l * HID * HID,
                                           wbf + W2_OFF + l * HID * HID,
                                           b1 + l * HID, b2 + l * HID,
                                           gam + l * HID, bet + l * HID);
  }
  out_proj<<<BN / 32, 256, 0, stream>>>(h, wbf + WO1_OFF, wbf + WO2_OFF,
                                        bo1, bo2, out);
}
